// Transformer_AE_61933428408755
// MI455X (gfx1250) — matrix operand order repaired and checked by reading, not compiled, not run on hardware
//
#include <hip/hip_runtime.h>
#include <hip/hip_bf16.h>
#include <math.h>

// ---------------------------------------------------------------------------
// Transformer_AE forward for MI455X (gfx1250, wave32, WMMA f32_16x16x32_f16)
// ---------------------------------------------------------------------------

typedef __attribute__((ext_vector_type(16))) _Float16 v16h;
typedef __attribute__((ext_vector_type(8)))  _Float16 v8h;
typedef __attribute__((ext_vector_type(4)))  _Float16 v4h;
typedef __attribute__((ext_vector_type(8)))  float    v8f;

#define TDIM   2048
#define BDIM   4
#define HDIM   512
#define NLAYER 4
#define DFFN   2048
#define MROWS  (BDIM * TDIM)
#define NHEADS 4
#define DHEAD  128
#define WINDOW 606

#define LPAD 40   // LDS row stride in halves (80B) -> 16B aligned, conflict-spread

__device__ __forceinline__ v8f wmma_f16(v16h a, v16h b, v8f c) {
  // v_wmma_f32_16x16x32_f16: D = A(16x32) * B(32x16) + C(16x16 f32)
  return __builtin_amdgcn_wmma_f32_16x16x32_f16(
      /*neg_a=*/false, a, /*neg_b=*/false, b,
      /*c_mod=*/(short)0, c, /*reuse_a=*/false, /*reuse_b=*/false);
}

__device__ __forceinline__ v16h cat8(v8h lo, v8h hi) {
  return __builtin_shufflevector(lo, hi, 0, 1, 2, 3, 4, 5, 6, 7,
                                         8, 9, 10, 11, 12, 13, 14, 15);
}

// convert two float4 to a packed v8h (optionally scaled)
__device__ __forceinline__ v8h cvt8(float4 a, float4 b, float s) {
  v8h r = { (_Float16)(a.x * s), (_Float16)(a.y * s),
            (_Float16)(a.z * s), (_Float16)(a.w * s),
            (_Float16)(b.x * s), (_Float16)(b.y * s),
            (_Float16)(b.z * s), (_Float16)(b.w * s) };
  return r;
}

__device__ __forceinline__ float gelu_exact(float x) {
  return 0.5f * x * (1.0f + erff(x * 0.70710678118654752f));
}

__device__ __forceinline__ float wave_sum32(float v) {
#pragma unroll
  for (int m = 16; m >= 1; m >>= 1) v += __shfl_xor(v, m, 32);
  return v;
}
__device__ __forceinline__ float half_max16(float v) {
#pragma unroll
  for (int m = 8; m >= 1; m >>= 1) v = fmaxf(v, __shfl_xor(v, m, 32));
  return v;
}
__device__ __forceinline__ float half_sum16(float v) {
#pragma unroll
  for (int m = 8; m >= 1; m >>= 1) v += __shfl_xor(v, m, 32);
  return v;
}

// ---------------------------------------------------------------------------
// Generic WMMA GEMM:  C[M,N] = act(A[M,K] @ W[K,N] + bias)
// flags: bit0 = add bias, bit1 = exact GELU
// Block: 128 threads = 4 waves; tile 64x64; BK=32 (one WMMA K-step).
// Staging is branch-free: float4 loads at clamped addresses + zero-select,
// so the compiler batches 8x global_load_b128 before a single wait, then
// converts fp32->f16 into padded LDS (B stored K-transposed so fragment
// fetches are contiguous ds_load_b128 in the ISA-required lane layouts).
// Requirements: M % 64 == 0 (M=8192 here); N,K multiples of 4 (ragged-safe).
// ---------------------------------------------------------------------------
#define GBM 64
#define GBN 64
#define GBK 32

__global__ __launch_bounds__(128) void gemm_wmma_kernel(
    const float* __restrict__ A, const float* __restrict__ W,
    const float* __restrict__ bias, float* __restrict__ C,
    int M, int N, int K, int flags)
{
  __shared__ alignas(16) _Float16 As[GBM * LPAD];  // [row][k]
  __shared__ alignas(16) _Float16 Bs[GBN * LPAD];  // [n][k] (K-transposed)

  const int tid  = threadIdx.x;
  const int wave = tid >> 5;
  const int lane = tid & 31;
  const int hs   = lane >> 4;   // half-wave select
  const int l16  = lane & 15;
  const int tileM = blockIdx.x * GBM;
  const int tileN = blockIdx.y * GBN;
  const int wm = (wave >> 1) * 32;
  const int wn = (wave & 1) * 32;

  v8f acc00 = {}, acc01 = {}, acc10 = {}, acc11 = {};
  const float4 zero4 = make_float4(0.0f, 0.0f, 0.0f, 0.0f);

  for (int k0 = 0; k0 < K; k0 += GBK) {
    // --- stage A tile (64x32): 4 x float4 per thread, branch-free ---
    float4 av[4];
#pragma unroll
    for (int it = 0; it < 4; ++it) {
      const int f = tid + 128 * it;           // float4 index in tile
      const int r = f >> 3;                   // 8 float4 per 32-wide row
      const int c4 = (f & 7) * 4;
      const int gk = k0 + c4;
      const int kc = (gk <= K - 4) ? gk : (K - 4);   // clamp: always valid
      av[it] = *(const float4*)(A + (size_t)(tileM + r) * K + kc);
    }
    // --- stage B tile (32x64): 4 x float4 per thread (coalesced) ---
    float4 bv[4];
#pragma unroll
    for (int it = 0; it < 4; ++it) {
      const int f = tid + 128 * it;
      const int kr = f >> 4;                  // 16 float4 per 64-wide row
      const int n4 = (f & 15) * 4;
      const int gk = k0 + kr;
      const int gn = tileN + n4;
      const int kc = (gk < K) ? gk : (K - 1);
      const int nc = (gn <= N - 4) ? gn : (N - 4);
      bv[it] = *(const float4*)(W + (size_t)kc * N + nc);
    }
    // speculative prefetch of next K-tile (global_prefetch_b8 on gfx1250)
    if (k0 + GBK < K) {
      __builtin_prefetch(&A[(size_t)(tileM + (tid >> 1)) * K + (k0 + GBK)], 0, 1);
      __builtin_prefetch(&W[(size_t)(k0 + GBK + (tid >> 2)) * N + tileN], 0, 1);
    }
    // zero-mask OOB chunks (selects, no branches), convert, store to LDS
#pragma unroll
    for (int it = 0; it < 4; ++it) {
      const int f = tid + 128 * it;
      const int r = f >> 3;
      const int c4 = (f & 7) * 4;
      float4 v = ((k0 + c4) < K) ? av[it] : zero4;
      v4h h = { (_Float16)v.x, (_Float16)v.y, (_Float16)v.z, (_Float16)v.w };
      *(v4h*)(&As[r * LPAD + c4]) = h;
    }
#pragma unroll
    for (int it = 0; it < 4; ++it) {
      const int f = tid + 128 * it;
      const int kr = f >> 4;
      const int n4 = (f & 15) * 4;
      const bool okc = ((k0 + kr) < K) && ((tileN + n4) < N);
      float4 v = okc ? bv[it] : zero4;
      Bs[(n4 + 0) * LPAD + kr] = (_Float16)v.x;
      Bs[(n4 + 1) * LPAD + kr] = (_Float16)v.y;
      Bs[(n4 + 2) * LPAD + kr] = (_Float16)v.z;
      Bs[(n4 + 3) * LPAD + kr] = (_Float16)v.w;
    }
    __syncthreads();

    // --- fragments ---
    // A 16-bit 16x32 layout: half-wave hs holds K {hs*8..+7} U {16+hs*8..+7}
    v16h a0, a1, b0, b1;
    {
      const _Float16* p = &As[(wm + l16) * LPAD];
      a0 = cat8(*(const v8h*)(p + hs * 8), *(const v8h*)(p + 16 + hs * 8));
      p = &As[(wm + 16 + l16) * LPAD];
      a1 = cat8(*(const v8h*)(p + hs * 8), *(const v8h*)(p + 16 + hs * 8));
      // B 16-bit 32x16 layout: half-wave hs holds K {hs*8..+7} U {16+hs*8..+7}
      const _Float16* q = &Bs[(wn + l16) * LPAD + hs * 8];
      b0 = cat8(*(const v8h*)q, *(const v8h*)(q + 16));
      q = &Bs[(wn + 16 + l16) * LPAD + hs * 8];
      b1 = cat8(*(const v8h*)q, *(const v8h*)(q + 16));
    }
    acc00 = wmma_f16(a0, b0, acc00);
    acc01 = wmma_f16(a0, b1, acc01);
    acc10 = wmma_f16(a1, b0, acc10);
    acc11 = wmma_f16(a1, b1, acc11);
    __syncthreads();
  }

  // --- epilogue: C/D layout: lane col = l16, row = v + 8*hs ---
#pragma unroll
  for (int i = 0; i < 2; ++i) {
#pragma unroll
    for (int j = 0; j < 2; ++j) {
      v8f a = (i == 0) ? (j == 0 ? acc00 : acc01) : (j == 0 ? acc10 : acc11);
      const int gcol = tileN + wn + j * 16 + l16;
      if (gcol < N) {
        const float bvs = (flags & 1) ? bias[gcol] : 0.0f;
#pragma unroll
        for (int v = 0; v < 8; ++v) {
          const int grow = tileM + wm + i * 16 + v + 8 * hs;
          float x = a[v] + bvs;
          if (flags & 2) x = gelu_exact(x);
          C[(size_t)grow * N + gcol] = x;
        }
      }
    }
  }
}

// ---------------------------------------------------------------------------
// Flash-style sliding-window attention, one wave per (batch, head, 16 q-rows)
// QKV layout: [B*T][3*HDIM]  (q | k | v), O layout: [B*T][HDIM]
// ---------------------------------------------------------------------------
__global__ __launch_bounds__(32) void attn_kernel(
    const float* __restrict__ QKV, float* __restrict__ O, int T)
{
  __shared__ alignas(16) _Float16 Pt[16 * LPAD];     // P tile, K padded 16->32
  __shared__ alignas(16) _Float16 Vt[DHEAD * LPAD];  // V^T [dh][key(32)]

  const int lane = threadIdx.x;
  const int hs = lane >> 4, l16 = lane & 15;
  const int qb   = blockIdx.x * 16;
  const int head = blockIdx.y;
  const int b    = blockIdx.z;
  const size_t rs = 3 * HDIM;
  const float* base = QKV + (size_t)b * T * rs;

  for (int i = lane; i < 16 * LPAD; i += 32) Pt[i] = (_Float16)0.0f;
  for (int i = lane; i < DHEAD * LPAD; i += 32) Vt[i] = (_Float16)0.0f;

  // Q fragments (A layout), pre-scaled by 1/sqrt(dh); vector loads
  const float qscale = 0.08838834764831845f;
  v16h qf[4];
  {
    const float* qp = base + (size_t)(qb + l16) * rs + head * DHEAD;
#pragma unroll
    for (int j = 0; j < 4; ++j) {
      const float4 q0 = *(const float4*)(qp + 32 * j + hs * 8);
      const float4 q1 = *(const float4*)(qp + 32 * j + hs * 8 + 4);
      const float4 q2 = *(const float4*)(qp + 32 * j + 16 + hs * 8);
      const float4 q3 = *(const float4*)(qp + 32 * j + 16 + hs * 8 + 4);
      qf[j] = cat8(cvt8(q0, q1, qscale), cvt8(q2, q3, qscale));
    }
  }

  float m_run[8], l_run[8];
#pragma unroll
  for (int v = 0; v < 8; ++v) { m_run[v] = -3.0e38f; l_run[v] = 0.0f; }
  v8f zf = {};
  v8f oacc[8];
#pragma unroll
  for (int t = 0; t < 8; ++t) oacc[t] = zf;

  int kmin = qb - (WINDOW - 1);
  if (kmin < 0) kmin = 0;
  for (int kb = (kmin & ~15); kb <= qb; kb += 16) {
    // ---- S = Q @ K^T over dh=128 (4 WMMA K-steps) ----
    v8f s = {};
    const float* kp = base + (size_t)(kb + l16) * rs + HDIM + head * DHEAD;
#pragma unroll
    for (int j = 0; j < 4; ++j) {
      // B layout: lane col = key l16, K = dh {hs*8..+7} U {16+hs*8..+7}
      const float4 k0 = *(const float4*)(kp + 32 * j + hs * 8);
      const float4 k1 = *(const float4*)(kp + 32 * j + hs * 8 + 4);
      const float4 k2 = *(const float4*)(kp + 32 * j + 16 + hs * 8);
      const float4 k3 = *(const float4*)(kp + 32 * j + 16 + hs * 8 + 4);
      const v16h kf = cat8(cvt8(k0, k1, 1.0f), cvt8(k2, k3, 1.0f));
      s = wmma_f16(qf[j], kf, s);
    }
    __syncthreads();

    // ---- mask + online softmax (rows spread over 16 lanes per half) ----
    const int key = kb + l16;
    float pv[8];
#pragma unroll
    for (int v = 0; v < 8; ++v) {
      const int qr = qb + v + 8 * hs;
      const bool ok = (key <= qr) && (key > qr - WINDOW);
      const float sv = ok ? s[v] : -1.0e30f;
      const float mx = half_max16(sv);
      const float mnew = fmaxf(m_run[v], mx);
      const float p = ok ? __expf(sv - mnew) : 0.0f;
      const float psum = half_sum16(p);
      const float scale = __expf(m_run[v] - mnew);
      l_run[v] = l_run[v] * scale + psum;
      m_run[v] = mnew;
      pv[v] = p;
#pragma unroll
      for (int t = 0; t < 8; ++t) oacc[t][v] = oacc[t][v] * scale;
    }

    // ---- stage P (C-layout -> LDS, cols 16..31 stay zero) ----
#pragma unroll
    for (int v = 0; v < 8; ++v)
      Pt[(v + 8 * hs) * LPAD + l16] = (_Float16)pv[v];

    // ---- stage V block transposed: Vt[dh][key]; float4 batches of 8 ----
    {
      const float* vb = base + (size_t)kb * rs + 2 * HDIM + head * DHEAD;
#pragma unroll
      for (int g = 0; g < 2; ++g) {
        float4 vv[8];
#pragma unroll
        for (int it = 0; it < 8; ++it) {
          const int f = lane + 32 * (g * 8 + it);  // float4 idx in 16x128 blk
          const int kk = f >> 5;                   // 32 float4 per key row
          const int d4 = (f & 31) * 4;
          vv[it] = *(const float4*)(vb + (size_t)kk * rs + d4);
        }
#pragma unroll
        for (int it = 0; it < 8; ++it) {
          const int f = lane + 32 * (g * 8 + it);
          const int kk = f >> 5;
          const int d4 = (f & 31) * 4;
          Vt[(d4 + 0) * LPAD + kk] = (_Float16)vv[it].x;
          Vt[(d4 + 1) * LPAD + kk] = (_Float16)vv[it].y;
          Vt[(d4 + 2) * LPAD + kk] = (_Float16)vv[it].z;
          Vt[(d4 + 3) * LPAD + kk] = (_Float16)vv[it].w;
        }
      }
    }
    __syncthreads();

    // ---- O += P @ V (8 dh tiles), K padded to 32 with zero P ----
    v16h pf;
    {
      const _Float16* pp = &Pt[l16 * LPAD];
      pf = cat8(*(const v8h*)(pp + hs * 8), *(const v8h*)(pp + 16 + hs * 8));
    }
#pragma unroll
    for (int nt = 0; nt < 8; ++nt) {
      const _Float16* vp = &Vt[(16 * nt + l16) * LPAD + hs * 8];
      v16h vf = cat8(*(const v8h*)vp, *(const v8h*)(vp + 16));
      oacc[nt] = wmma_f16(pf, vf, oacc[nt]);
    }
    __syncthreads();
  }

  // ---- normalize and store ----
#pragma unroll
  for (int nt = 0; nt < 8; ++nt) {
    const int gcol = head * DHEAD + 16 * nt + l16;
#pragma unroll
    for (int v = 0; v < 8; ++v) {
      const int grow = qb + v + 8 * hs;
      const float o = oacc[nt][v] / fmaxf(l_run[v], 1e-20f);
      O[((size_t)b * T + grow) * HDIM + gcol] = o;
    }
  }
}

// ---------------------------------------------------------------------------
// LayerNorm (optionally with residual): out = LN(X (+R)) * g + b
// One wave32 per 512-wide row; 8 rows per 256-thread block.
// ---------------------------------------------------------------------------
__global__ __launch_bounds__(256) void ln_kernel(
    const float* __restrict__ X, const float* __restrict__ R,
    const float* __restrict__ g, const float* __restrict__ be,
    float* __restrict__ out, int hasR)
{
  const int wave = threadIdx.x >> 5, lane = threadIdx.x & 31;
  const int row = blockIdx.x * 8 + wave;
  const float* xr = X + (size_t)row * HDIM;
  const float* rr = R + (size_t)row * HDIM;
  float vals[16];
  float s = 0.0f;
#pragma unroll
  for (int i = 0; i < 16; ++i) {
    const int c = lane + 32 * i;
    float v = xr[c];
    if (hasR) v += rr[c];
    vals[i] = v;
    s += v;
  }
  s = wave_sum32(s);
  const float mean = s * (1.0f / HDIM);
  float vs = 0.0f;
#pragma unroll
  for (int i = 0; i < 16; ++i) { const float d = vals[i] - mean; vs += d * d; }
  vs = wave_sum32(vs);
  const float rstd = rsqrtf(vs * (1.0f / HDIM) + 1e-5f);
  float* orow = out + (size_t)row * HDIM;
#pragma unroll
  for (int i = 0; i < 16; ++i) {
    const int c = lane + 32 * i;
    orow[c] = (vals[i] - mean) * rstd * g[c] + be[c];
  }
}

// ---------------------------------------------------------------------------
// Input embedding: X = u_aug @ W_in + b_in + pos_encoding
// (y_aug is zero; y_true peek only for t < 30)
// ---------------------------------------------------------------------------
__global__ __launch_bounds__(256) void embed_kernel(
    const float* __restrict__ u, const float* __restrict__ y,
    const float* __restrict__ Win, const float* __restrict__ bin,
    float* __restrict__ X)
{
  const int idx = blockIdx.x * 256 + threadIdx.x;
  if (idx >= MROWS * HDIM) return;
  const int j = idx & (HDIM - 1);
  const int m = idx >> 9;
  const int b = m / TDIM, t = m % TDIM;
  float acc = bin[j];
  const float* ub = u + (size_t)b * 8 * TDIM;
#pragma unroll
  for (int i = 0; i < 8; ++i) acc += ub[i * TDIM + t] * Win[i * HDIM + j];
  if (t < 30) {
    const float* yb = y + (size_t)b * 8 * TDIM;
#pragma unroll
    for (int i = 0; i < 8; ++i) acc += yb[i * TDIM + t] * Win[(16 + i) * HDIM + j];
  }
  const int p2 = (j >> 1) * 2;
  const float div = __expf(-(float)p2 * (9.210340371976184f / (float)HDIM));
  const float ang = (float)t * div;
  acc += (j & 1) ? __cosf(ang) : __sinf(ang);
  X[idx] = acc;
}

// ---------------------------------------------------------------------------
// Loss: sum((y_hat - y_t)^2); y is [B, 8, T]; y_hat is [B*T, 8]
// ---------------------------------------------------------------------------
__global__ void zero_kernel(float* out) {
  if (threadIdx.x == 0 && blockIdx.x == 0) out[0] = 0.0f;
}

__global__ __launch_bounds__(256) void loss_kernel(
    const float* __restrict__ yh, const float* __restrict__ y,
    float* __restrict__ out)
{
  __shared__ float red[8];
  const int idx = blockIdx.x * 256 + threadIdx.x;
  float d2 = 0.0f;
  if (idx < MROWS * 8) {
    const int dcol = idx & 7, m = idx >> 3;
    const int b = m / TDIM, t = m % TDIM;
    const float yt = y[((size_t)b * 8 + dcol) * TDIM + t];
    const float d = yh[idx] - yt;
    d2 = d * d;
  }
  d2 = wave_sum32(d2);
  const int wave = threadIdx.x >> 5, lane = threadIdx.x & 31;
  if (lane == 0) red[wave] = d2;
  __syncthreads();
  if (threadIdx.x == 0) {
    float s = 0.0f;
    for (int w = 0; w < 8; ++w) s += red[w];
    atomicAdd(out, s);
  }
}

// ---------------------------------------------------------------------------
extern "C" void kernel_launch(void* const* d_in, const int* in_sizes, int n_in,
                              void* d_out, int out_size, void* d_ws, size_t ws_size,
                              hipStream_t stream) {
  (void)in_sizes; (void)n_in; (void)out_size; (void)ws_size;
  const float* u     = (const float*)d_in[0];
  const float* y     = (const float*)d_in[1];
  const float* W_in  = (const float*)d_in[2];
  const float* b_in  = (const float*)d_in[3];
  const float* qkv_w = (const float*)d_in[4];
  const float* qkv_b = (const float*)d_in[5];
  const float* out_w = (const float*)d_in[6];
  const float* out_b = (const float*)d_in[7];
  const float* ff1_w = (const float*)d_in[8];
  const float* ff1_b = (const float*)d_in[9];
  const float* ff2_w = (const float*)d_in[10];
  const float* ff2_b = (const float*)d_in[11];
  const float* ln1_g = (const float*)d_in[12];
  const float* ln1_b = (const float*)d_in[13];
  const float* ln2_g = (const float*)d_in[14];
  const float* ln2_b = (const float*)d_in[15];
  const float* lnf_g = (const float*)d_in[16];
  const float* lnf_b = (const float*)d_in[17];
  const float* xm1_w = (const float*)d_in[18];
  const float* xm1_b = (const float*)d_in[19];
  const float* xm2_w = (const float*)d_in[20];
  const float* xm2_b = (const float*)d_in[21];
  const float* me1_w = (const float*)d_in[22];
  const float* me1_b = (const float*)d_in[23];
  const float* me2_w = (const float*)d_in[24];
  const float* me2_b = (const float*)d_in[25];
  const float* me3_w = (const float*)d_in[26];
  const float* me3_b = (const float*)d_in[27];
  float* outp = (float*)d_out;

  // workspace: X (16MB) | TMP (16MB) | BIG (64MB)  => 96MB total
  float* X   = (float*)d_ws;
  float* TMP = X + (size_t)MROWS * HDIM;
  float* BIG = TMP + (size_t)MROWS * HDIM;

  const dim3 blk128(128), blk256(256), blk32(32);

  embed_kernel<<<(MROWS * HDIM + 255) / 256, blk256, 0, stream>>>(u, y, W_in, b_in, X);

  for (int l = 0; l < NLAYER; ++l) {
    // QKV = X @ Wqkv + b        [M, 1536]
    gemm_wmma_kernel<<<dim3(MROWS / 64, (3 * HDIM) / 64), blk128, 0, stream>>>(
        X, qkv_w + (size_t)l * HDIM * 3 * HDIM, qkv_b + (size_t)l * 3 * HDIM, BIG,
        MROWS, 3 * HDIM, HDIM, 1);
    // ATT = softmax(QK^T/sqrt(dh)) V   [M, 512]
    attn_kernel<<<dim3(TDIM / 16, NHEADS, BDIM), blk32, 0, stream>>>(BIG, TMP, TDIM);
    // PROJ = ATT @ Wo + b
    gemm_wmma_kernel<<<dim3(MROWS / 64, HDIM / 64), blk128, 0, stream>>>(
        TMP, out_w + (size_t)l * HDIM * HDIM, out_b + (size_t)l * HDIM, BIG,
        MROWS, HDIM, HDIM, 1);
    // X = LN(X + PROJ)
    ln_kernel<<<MROWS / 8, blk256, 0, stream>>>(
        X, BIG, ln1_g + l * HDIM, ln1_b + l * HDIM, X, 1);
    // FF1 = gelu(X @ W1 + b)    [M, 2048]
    gemm_wmma_kernel<<<dim3(MROWS / 64, DFFN / 64), blk128, 0, stream>>>(
        X, ff1_w + (size_t)l * HDIM * DFFN, ff1_b + (size_t)l * DFFN, BIG,
        MROWS, DFFN, HDIM, 3);
    // FF2 = FF1 @ W2 + b        [M, 512]
    gemm_wmma_kernel<<<dim3(MROWS / 64, HDIM / 64), blk128, 0, stream>>>(
        BIG, ff2_w + (size_t)l * DFFN * HDIM, ff2_b + (size_t)l * HDIM, TMP,
        MROWS, HDIM, DFFN, 1);
    // X = LN(X + FF2)
    ln_kernel<<<MROWS / 8, blk256, 0, stream>>>(
        X, TMP, ln2_g + l * HDIM, ln2_b + l * HDIM, X, 1);
  }
  // final LN (in place, no residual)
  ln_kernel<<<MROWS / 8, blk256, 0, stream>>>(X, X, lnf_g, lnf_b, X, 0);

  // head MLPs
  gemm_wmma_kernel<<<dim3(MROWS / 64, HDIM / 64), blk128, 0, stream>>>(
      X, xm1_w, xm1_b, TMP, MROWS, HDIM, HDIM, 3);               // gelu
  gemm_wmma_kernel<<<dim3(MROWS / 64, 1), blk128, 0, stream>>>(
      TMP, xm2_w, xm2_b, BIG, MROWS, 16, HDIM, 1);               // z [M,16]
  gemm_wmma_kernel<<<dim3(MROWS / 64, HDIM / 64), blk128, 0, stream>>>(
      BIG, me1_w, me1_b, TMP, MROWS, HDIM, 16, 3);               // gelu, K=16
  gemm_wmma_kernel<<<dim3(MROWS / 64, HDIM / 64), blk128, 0, stream>>>(
      TMP, me2_w, me2_b, BIG, MROWS, HDIM, HDIM, 3);             // gelu
  gemm_wmma_kernel<<<dim3(MROWS / 64, 1), blk128, 0, stream>>>(
      BIG, me3_w, me3_b, TMP, MROWS, 8, HDIM, 1);                // y_hat [M,8]

  zero_kernel<<<1, blk32, 0, stream>>>(outp);
  loss_kernel<<<(MROWS * 8 + 255) / 256, blk256, 0, stream>>>(TMP, y, outp);
}
